// LinearAttention_61778809585929
// MI455X (gfx1250) — matrix operand order repaired and checked by reading, not compiled, not run on hardware
//
#include <hip/hip_runtime.h>

// ---------------------------------------------------------------------------
// LinearAttention on MI455X (gfx1250): bf16 WMMA pipeline, f32 accumulation,
// async-to-LDS double-buffered GEMM staging.
//   B=8, N=4096, D_IN=D_OUT=1024, M = B*N = 32768
// ---------------------------------------------------------------------------

#define BB     8
#define NSEQ   4096
#define DD     1024
#define MTOT   (BB * NSEQ)   // 32768

typedef __attribute__((ext_vector_type(16))) __bf16 v16bf;
typedef __attribute__((ext_vector_type(8)))  float  v8f;

union FragBF {
    v16bf v;
    uint4 u[2];
};

// fp32 -> bf16 with round-to-nearest-even
static __device__ __forceinline__ unsigned short f2bf(float f) {
    unsigned int u = __float_as_uint(f);
    u += 0x7FFFu + ((u >> 16) & 1u);
    return (unsigned short)(u >> 16);
}

// ---------------------------------------------------------------------------
// CDNA5 async copy: global -> LDS, 16 bytes per lane, tracked by ASYNCcnt.
// lds_off = LDS byte address (low 32 bits of flat shared pointer).
// ---------------------------------------------------------------------------
static __device__ __forceinline__ void async_b128(unsigned lds_off,
                                                  const unsigned short* g) {
    asm volatile("global_load_async_to_lds_b128 %0, %1, off"
                 :: "v"(lds_off), "v"(g) : "memory");
}
static __device__ __forceinline__ void wait_async_le4() {
    asm volatile("s_wait_asynccnt 0x4" ::: "memory");
}
static __device__ __forceinline__ void wait_async_0() {
    asm volatile("s_wait_asynccnt 0x0" ::: "memory");
}

// ---------------------------------------------------------------------------
// fp32 -> bf16 conversion kernel, vectorized (4 elems/thread/iter)
// ---------------------------------------------------------------------------
__global__ void cvt_bf16(const float4* __restrict__ src,
                         uint2* __restrict__ dst, int n4) {
    for (int i = blockIdx.x * blockDim.x + threadIdx.x; i < n4;
         i += gridDim.x * blockDim.x) {
        float4 f = src[i];
        uint2 o;
        o.x = (unsigned)f2bf(f.x) | ((unsigned)f2bf(f.y) << 16);
        o.y = (unsigned)f2bf(f.z) | ((unsigned)f2bf(f.w) << 16);
        dst[i] = o;
    }
}

// ---------------------------------------------------------------------------
// Block GEMM core: C = A * B^T, block tile 128(M) x 128(N), K stages of 32.
//   A : rows m_blk..+127, K-contiguous, stride lda (elements)
//   Bm: rows n_blk..+127 of B^T, K-contiguous, stride ldb
// 8 waves (4 M x 2 N), wave tile 32x64 = 2x4 WMMA accumulators.
// A/B K-slices staged in LDS via global_load_async_to_lds_b128, double buffer.
// Fragment layouts per CDNA5 ISA 7.12.2 (A 16x32, B 32x16 bf16).
// ---------------------------------------------------------------------------
static __device__ __forceinline__ void block_gemm_128x128(
    const unsigned short* __restrict__ A, int lda,
    const unsigned short* __restrict__ Bm, int ldb,
    int m_blk, int n_blk, int Kdim,
    unsigned short sA[][128 * 32], unsigned short sB[][128 * 32],
    v8f acc[2][4]) {

    const int t    = threadIdx.x;
    const int lane = t & 31;
    const int half = lane >> 4;
    const int l15  = lane & 15;
    const int wave = t >> 5;
    const int wm   = wave >> 1;   // 0..3
    const int wn   = wave & 1;    // 0..1

    // loader assignment: thread t copies 32B of A and 32B of B per stage
    const int arow = t >> 1;            // 0..127
    const int achk = (t & 1) * 16;      // element offset within 32-elem row
    const unsigned short* ga = A  + (size_t)(m_blk + arow) * lda + achk;
    const unsigned short* gb = Bm + (size_t)(n_blk + arow) * ldb + achk;

    const v8f z = {0.f, 0.f, 0.f, 0.f, 0.f, 0.f, 0.f, 0.f};
    #pragma unroll
    for (int i = 0; i < 2; ++i)
        #pragma unroll
        for (int j = 0; j < 4; ++j) acc[i][j] = z;

    const int S = Kdim >> 5;   // stages of K=32

    auto issue_stage = [&](int s) {
        const int buf = s & 1;
        const unsigned short* pa = ga + (size_t)s * 32;
        const unsigned short* pb = gb + (size_t)s * 32;
        const unsigned la = (unsigned)(size_t)&sA[buf][arow * 32 + achk];
        const unsigned lb = (unsigned)(size_t)&sB[buf][arow * 32 + achk];
        async_b128(la,      pa);       // 4 async instrs/wave/stage -> cnt +4
        async_b128(la + 16, pa + 8);
        async_b128(lb,      pb);
        async_b128(lb + 16, pb + 8);
    };

    issue_stage(0);
    if (S > 1) issue_stage(1);

    for (int s = 0; s < S; ++s) {
        // drain this stage's copies (in-order: <=4 leaves only stage s+1)
        if (s + 1 < S) wait_async_le4();
        else           wait_async_0();
        __syncthreads();

        const int cb = s & 1;
        FragBF a[2], b[4];
        #pragma unroll
        for (int i = 0; i < 2; ++i) {
            const int row = wm * 32 + 16 * i + l15;
            const unsigned short* p = &sA[cb][row * 32 + half * 8];
            a[i].u[0] = *(const uint4*)p;
            a[i].u[1] = *(const uint4*)(p + 16);
        }
        #pragma unroll
        for (int j = 0; j < 4; ++j) {
            const int col = wn * 64 + 16 * j + l15;
            const unsigned short* p = &sB[cb][col * 32 + half * 8];
            b[j].u[0] = *(const uint4*)p;
            b[j].u[1] = *(const uint4*)(p + 16);
        }
        #pragma unroll
        for (int i = 0; i < 2; ++i)
            #pragma unroll
            for (int j = 0; j < 4; ++j)
                acc[i][j] = __builtin_amdgcn_wmma_f32_16x16x32_bf16(
                    false, a[i].v, false, b[j].v, (short)0, acc[i][j],
                    false, false);

        __syncthreads();               // all waves done reading buf (s&1)
        if (s + 2 < S) issue_stage(s + 2);
    }
}

// ---------------------------------------------------------------------------
// QKV projection: C[m, e] = sum_k x[m,k] * W[e,k] + bias[e]
//   wsel 0/1 -> transposed bf16 [B, D, N]; wsel 2 -> bf16 [B*N, D]
// ---------------------------------------------------------------------------
__global__ void gemm_qkv(const unsigned short* __restrict__ xb,
                         const unsigned short* __restrict__ wb,
                         const float* __restrict__ bq,
                         const float* __restrict__ bk,
                         const float* __restrict__ bv,
                         unsigned short* __restrict__ qt,
                         unsigned short* __restrict__ kt,
                         unsigned short* __restrict__ vn) {
    __shared__ unsigned short sA[2][128 * 32];
    __shared__ unsigned short sB[2][128 * 32];

    const int lane = threadIdx.x & 31;
    const int half = lane >> 4;
    const int l15  = lane & 15;
    const int wave = threadIdx.x >> 5;
    const int wm = wave >> 1, wn = wave & 1;

    const int wsel = blockIdx.z;
    const unsigned short* W = wb + (size_t)wsel * DD * DD;
    const float* bias = (wsel == 0) ? bq : ((wsel == 1) ? bk : bv);

    const int m_blk = blockIdx.y * 128;
    const int n_blk = blockIdx.x * 128;

    v8f acc[2][4];
    block_gemm_128x128(xb, DD, W, DD, m_blk, n_blk, DD, sA, sB, acc);

    unsigned short* tdst = (wsel == 0) ? qt : kt;
    #pragma unroll
    for (int i = 0; i < 2; ++i) {
        const int rowb = m_blk + wm * 32 + 16 * i + half * 8;
        #pragma unroll
        for (int j = 0; j < 4; ++j) {
            const int col = n_blk + wn * 64 + 16 * j + l15;
            const float bcol = bias[col];
            #pragma unroll
            for (int r = 0; r < 8; ++r) {
                const int m = rowb + r;
                const unsigned short h = f2bf(acc[i][j][r] + bcol);
                if (wsel == 2) {
                    vn[(size_t)m * DD + col] = h;
                } else {
                    const int b  = m >> 12;        // m / NSEQ
                    const int nn = m & (NSEQ - 1); // m % NSEQ
                    tdst[((size_t)b * DD + col) * NSEQ + nn] = h;
                }
            }
        }
    }
}

// ---------------------------------------------------------------------------
// Attention scores: S[b, e, f] = sum_n Q[b,n,e] * K[b,n,f]  (Kdim = NSEQ)
// ---------------------------------------------------------------------------
__global__ void gemm_attn(const unsigned short* __restrict__ qt,
                          const unsigned short* __restrict__ kt,
                          float* __restrict__ attn) {
    __shared__ unsigned short sA[2][128 * 32];
    __shared__ unsigned short sB[2][128 * 32];

    const int lane = threadIdx.x & 31;
    const int half = lane >> 4;
    const int l15  = lane & 15;
    const int wave = threadIdx.x >> 5;
    const int wm = wave >> 1, wn = wave & 1;
    const int b = blockIdx.z;

    const unsigned short* A  = qt + (size_t)b * DD * NSEQ;
    const unsigned short* Bm = kt + (size_t)b * DD * NSEQ;
    float* out = attn + (size_t)b * DD * DD;

    const int m_blk = blockIdx.y * 128;
    const int n_blk = blockIdx.x * 128;

    v8f acc[2][4];
    block_gemm_128x128(A, NSEQ, Bm, NSEQ, m_blk, n_blk, NSEQ, sA, sB, acc);

    #pragma unroll
    for (int i = 0; i < 2; ++i) {
        const int rowb = m_blk + wm * 32 + 16 * i + half * 8;
        #pragma unroll
        for (int j = 0; j < 4; ++j) {
            const int col = n_blk + wn * 64 + 16 * j + l15;
            #pragma unroll
            for (int r = 0; r < 8; ++r)
                out[(size_t)(rowb + r) * DD + col] = acc[i][j][r];
        }
    }
}

// ---------------------------------------------------------------------------
// Row softmax over S[b, e, :] + transposed bf16 write: Pt[b, f, e]
// ---------------------------------------------------------------------------
__global__ void softmax_t(const float* __restrict__ attn,
                          unsigned short* __restrict__ pt) {
    __shared__ float red[256];
    const int row = blockIdx.x;          // b*D + e
    const int b = row >> 10;
    const int e = row & (DD - 1);
    const float* src = attn + (size_t)row * DD;
    const int t = threadIdx.x;

    float vals[4];
    float m = -3.402823466e+38f;
    #pragma unroll
    for (int i = 0; i < 4; ++i) {
        vals[i] = src[t + 256 * i];
        m = fmaxf(m, vals[i]);
    }
    red[t] = m;
    __syncthreads();
    for (int s = 128; s > 0; s >>= 1) {
        if (t < s) red[t] = fmaxf(red[t], red[t + s]);
        __syncthreads();
    }
    m = red[0];
    __syncthreads();

    float sum = 0.f;
    #pragma unroll
    for (int i = 0; i < 4; ++i) {
        vals[i] = __expf(vals[i] - m);
        sum += vals[i];
    }
    red[t] = sum;
    __syncthreads();
    for (int s = 128; s > 0; s >>= 1) {
        if (t < s) red[t] += red[t + s];
        __syncthreads();
    }
    const float inv = 1.0f / red[0];

    unsigned short* dst = pt + (size_t)b * DD * DD + e;   // [b, f, e]
    #pragma unroll
    for (int i = 0; i < 4; ++i)
        dst[(size_t)(t + 256 * i) * DD] = f2bf(vals[i] * inv);
}

// ---------------------------------------------------------------------------
// Output: out[b, n, f] = sum_e V[b,n,e] * P[b,e,f]   (P stored transposed)
// ---------------------------------------------------------------------------
__global__ void gemm_out(const unsigned short* __restrict__ vn,
                         const unsigned short* __restrict__ pt,
                         float* __restrict__ out) {
    __shared__ unsigned short sA[2][128 * 32];
    __shared__ unsigned short sB[2][128 * 32];

    const int lane = threadIdx.x & 31;
    const int half = lane >> 4;
    const int l15  = lane & 15;
    const int wave = threadIdx.x >> 5;
    const int wm = wave >> 1, wn = wave & 1;
    const int b = blockIdx.z;

    const unsigned short* A  = vn + (size_t)b * NSEQ * DD;
    const unsigned short* Bm = pt + (size_t)b * DD * DD;
    float* dst = out + (size_t)b * NSEQ * DD;

    const int m_blk = blockIdx.y * 128;
    const int n_blk = blockIdx.x * 128;

    v8f acc[2][4];
    block_gemm_128x128(A, DD, Bm, DD, m_blk, n_blk, DD, sA, sB, acc);

    #pragma unroll
    for (int i = 0; i < 2; ++i) {
        const int rowb = m_blk + wm * 32 + 16 * i + half * 8;
        #pragma unroll
        for (int j = 0; j < 4; ++j) {
            const int col = n_blk + wn * 64 + 16 * j + l15;
            #pragma unroll
            for (int r = 0; r < 8; ++r)
                dst[(size_t)(rowb + r) * DD + col] = acc[i][j][r];
        }
    }
}

// ---------------------------------------------------------------------------
// Host launch
// ---------------------------------------------------------------------------
extern "C" void kernel_launch(void* const* d_in, const int* in_sizes, int n_in,
                              void* d_out, int out_size, void* d_ws, size_t ws_size,
                              hipStream_t stream) {
    const float* x  = (const float*)d_in[0];
    const float* Wq = (const float*)d_in[1];
    const float* bq = (const float*)d_in[2];
    const float* Wk = (const float*)d_in[3];
    const float* bk = (const float*)d_in[4];
    const float* Wv = (const float*)d_in[5];
    const float* bv = (const float*)d_in[6];
    float* out = (float*)d_out;

    // Workspace layout (bf16 stored as ushort)
    unsigned short* xb = (unsigned short*)d_ws;                 // [M, D]    64 MB
    unsigned short* wb = xb + (size_t)MTOT * DD;                // [3, D, D]  6 MB
    unsigned short* qt = wb + (size_t)3 * DD * DD;              // [B, D, N] 64 MB
    unsigned short* kt = qt + (size_t)BB * DD * NSEQ;           // [B, D, N] 64 MB
    unsigned short* vn = kt + (size_t)BB * DD * NSEQ;           // [B, N, D] 64 MB
    float* attn        = (float*)(vn + (size_t)BB * NSEQ * DD); // [B, D, D] 32 MB
    unsigned short* pt = (unsigned short*)(attn + (size_t)BB * DD * DD); // 16 MB

    // 1) fp32 -> bf16
    cvt_bf16<<<2048, 256, 0, stream>>>((const float4*)x, (uint2*)xb,
                                       (MTOT / 4) * DD);
    cvt_bf16<<<512, 256, 0, stream>>>((const float4*)Wq, (uint2*)wb,
                                      DD * DD / 4);
    cvt_bf16<<<512, 256, 0, stream>>>((const float4*)Wk,
                                      (uint2*)(wb + (size_t)DD * DD), DD * DD / 4);
    cvt_bf16<<<512, 256, 0, stream>>>((const float4*)Wv,
                                      (uint2*)(wb + (size_t)2 * DD * DD), DD * DD / 4);

    // 2) QKV projections (q,k transposed; v row-major)
    gemm_qkv<<<dim3(DD / 128, MTOT / 128, 3), 256, 0, stream>>>(
        xb, wb, bq, bk, bv, qt, kt, vn);

    // 3) attention scores per batch: [D, D] = Q^T K over NSEQ
    gemm_attn<<<dim3(DD / 128, DD / 128, BB), 256, 0, stream>>>(qt, kt, attn);

    // 4) softmax rows -> transposed bf16 P^T
    softmax_t<<<BB * DD, 256, 0, stream>>>(attn, pt);

    // 5) out = V * P per batch
    gemm_out<<<dim3(DD / 128, NSEQ / 128, BB), 256, 0, stream>>>(vn, pt, out);
}
